// NeighborEmbedding_own_67997922231113
// MI455X (gfx1250) — matrix operand order repaired and checked by reading, not compiled, not run on hardware
//
#include <hip/hip_runtime.h>

typedef _Float16 h16;
typedef _Float16 v16h __attribute__((ext_vector_type(16)));
typedef _Float16 v8h  __attribute__((ext_vector_type(8)));
typedef float    v8f  __attribute__((ext_vector_type(8)));

static constexpr int cB  = 16;    // batch
static constexpr int cN  = 600;   // points
static constexpr int NP0 = 640;   // stage-0 padded points (mult of 64)
static constexpr int NP2 = 64;    // stage-2 padded points

// ---------------------------------------------------------------- utilities
__global__ void k_zero_f32(float* p, int n) {
  int i = blockIdx.x * blockDim.x + threadIdx.x;
  if (i < n) p[i] = 0.f;
}

__global__ void k_cvt_f16(const float* __restrict__ s, h16* __restrict__ d, int n) {
  int i = blockIdx.x * blockDim.x + threadIdx.x;
  if (i < n) d[i] = (h16)s[i];
}

// x [b][3][600] -> xyz [b][600][3]
__global__ void k_xyz(const float* __restrict__ x, float* __restrict__ xyz) {
  int i = blockIdx.x * blockDim.x + threadIdx.x;
  if (i >= cB * cN * 3) return;
  int b = i / (cN * 3), r = i % (cN * 3), n = r / 3, j = r % 3;
  xyz[i] = x[(b * 3 + j) * cN + n];
}

// ---------------------------------------------------------------- embedding
__global__ void k_embed1(const float* __restrict__ x, const float* __restrict__ w,
                         float* __restrict__ raw) {   // grid cB*cN, block 64
  int bn = blockIdx.x; int o = threadIdx.x;
  int b = bn / cN, n = bn % cN;
  const float* xp = x + (long)(b * 3) * cN + n;
  raw[(long)bn * 64 + o] = w[o*3]*xp[0] + w[o*3+1]*xp[cN] + w[o*3+2]*xp[2*cN];
}

__global__ void k_embed2(const h16* __restrict__ f1, const float* __restrict__ w,
                         float* __restrict__ raw) {   // grid cB*cN, block 128
  int bn = blockIdx.x; int o = threadIdx.x;
  const h16* fp = f1 + (long)bn * 64;
  float a = 0.f;
  #pragma unroll
  for (int c = 0; c < 64; ++c) a += w[o*64 + c] * (float)fp[c];
  raw[(long)bn * 128 + o] = a;
}

// ------------------------------------------------------------ batch-norm stats
// layout: [batch][row][C]; accumulate sum / sumsq per channel. blockDim == C.
template <typename T>
__global__ void k_stats(const T* __restrict__ buf, int C, int rowsPB, long bStride,
                        int rowStride, float* __restrict__ acc, long totalRows) {
  int c = threadIdx.x;
  long r0 = (long)blockIdx.x * 128;
  long r1 = r0 + 128 < totalRows ? r0 + 128 : totalRows;
  float s = 0.f, q = 0.f;
  for (long r = r0; r < r1; ++r) {
    long b = r / rowsPB, rr = r % rowsPB;
    float v = (float)buf[b * bStride + rr * (long)rowStride + c];
    s += v; q += v * v;
  }
  atomicAdd(&acc[c], s);
  atomicAdd(&acc[C + c], q);
}

__global__ void k_bnparams(const float* __restrict__ acc, const float* __restrict__ g,
                           const float* __restrict__ b, float cnt,
                           float* __restrict__ scale, float* __restrict__ shift, int C) {
  int c = threadIdx.x;
  if (c >= C) return;
  float m  = acc[c] / cnt;
  float v  = acc[C + c] / cnt - m * m;
  float sc = g[c] * rsqrtf(v + 1e-5f);
  scale[c] = sc;
  shift[c] = b[c] - m * sc;
}

__global__ void k_bnrelu_f32_f16(const float* __restrict__ in, h16* __restrict__ out,
                                 const float* __restrict__ scale, const float* __restrict__ shift,
                                 int C, long n) {
  long i = (long)blockIdx.x * blockDim.x + threadIdx.x;
  if (i >= n) return;
  int c = (int)(i % C);
  out[i] = (h16)fmaxf(in[i] * scale[c] + shift[c], 0.f);
}

__global__ void k_bnrelu_f16(h16* buf, const float* __restrict__ scale,
                             const float* __restrict__ shift, int C, long n) {
  long i = (long)blockIdx.x * blockDim.x + threadIdx.x;
  if (i >= n) return;
  int c = (int)(i % C);
  float v = (float)buf[i] * scale[c] + shift[c];
  buf[i] = (h16)fmaxf(v, 0.f);
}

// --------------------------------------------------------------- sampling
__global__ void k_fps(const float* __restrict__ coords, int n, int s, int* __restrict__ out) {
  // one block per batch, block 256. coords [b][n][3]
  int b = blockIdx.x, t = threadIdx.x;
  const float* cp = coords + (long)b * n * 3;
  __shared__ float dist[640];
  __shared__ float rv[256];
  __shared__ int   ri[256];
  __shared__ float cx, cy, cz;
  for (int i = t; i < n; i += 256) dist[i] = 1e10f;
  int far = 0;
  __syncthreads();
  for (int it = 0; it < s; ++it) {
    if (t == 0) { out[b*s + it] = far; cx = cp[far*3]; cy = cp[far*3+1]; cz = cp[far*3+2]; }
    __syncthreads();
    float bv = -1.f; int bi = 0;
    for (int i = t; i < n; i += 256) {
      float dx = cp[i*3] - cx, dy = cp[i*3+1] - cy, dz = cp[i*3+2] - cz;
      float d = dx*dx + dy*dy + dz*dz;
      float dm = fminf(dist[i], d);
      dist[i] = dm;
      if (dm > bv) { bv = dm; bi = i; }
    }
    rv[t] = bv; ri[t] = bi;
    __syncthreads();
    for (int off = 128; off; off >>= 1) {
      if (t < off) {
        if (rv[t+off] > rv[t] || (rv[t+off] == rv[t] && ri[t+off] < ri[t])) {
          rv[t] = rv[t+off]; ri[t] = ri[t+off];
        }
      }
      __syncthreads();
    }
    far = ri[0];
    __syncthreads();
  }
}

__global__ void k_knn(const float* __restrict__ coords, int n, const int* __restrict__ fidx,
                      int s, int k, int* __restrict__ nidx) {
  // one block per (b,si), block 256
  int b = blockIdx.x / s, si = blockIdx.x % s, t = threadIdx.x;
  const float* cp = coords + (long)b * n * 3;
  __shared__ float d[640];
  __shared__ float rv[256];
  __shared__ int   ri[256];
  int qi = fidx[b*s + si];
  float qx = cp[qi*3], qy = cp[qi*3+1], qz = cp[qi*3+2];
  for (int i = t; i < n; i += 256) {
    float dx = cp[i*3] - qx, dy = cp[i*3+1] - qy, dz = cp[i*3+2] - qz;
    d[i] = dx*dx + dy*dy + dz*dz;
  }
  __syncthreads();
  for (int j = 0; j < k; ++j) {
    float bv = 1e30f; int bi = 0;
    for (int i = t; i < n; i += 256) if (d[i] < bv) { bv = d[i]; bi = i; }
    rv[t] = bv; ri[t] = bi;
    __syncthreads();
    for (int off = 128; off; off >>= 1) {
      if (t < off) {
        if (rv[t+off] < rv[t] || (rv[t+off] == rv[t] && ri[t+off] < ri[t])) {
          rv[t] = rv[t+off]; ri[t] = ri[t+off];
        }
      }
      __syncthreads();
    }
    if (t == 0) { nidx[((long)b*s + si)*k + j] = ri[0]; d[ri[0]] = 1e30f; }
    __syncthreads();
  }
}

__global__ void k_gather3(const float* __restrict__ coords, const int* __restrict__ idx,
                          int s, int n, float* __restrict__ out) {
  int i = blockIdx.x * blockDim.x + threadIdx.x;
  if (i >= cB * s * 3) return;
  int b = i / (s*3), r = i % (s*3), si = r / 3, j = r % 3;
  out[i] = coords[((long)b*n + idx[b*s + si])*3 + j];
}

// agg column build: col = (b*s+si)*k+kj ; out [col][2D] = [nbr-ctr ; ctr]
__global__ void k_agg(const h16* __restrict__ feat, long featBS,
                      const int* __restrict__ fidx, const int* __restrict__ nidx,
                      int s, int k, int D, h16* __restrict__ agg) {  // blockDim = D
  long col = blockIdx.x;
  int kj = (int)(col % k);
  long g = col / k;
  int si = (int)(g % s), b = (int)(g / s);
  int cen = fidx[b*s + si];
  int nid = nidx[((long)b*s + si)*k + kj];
  const h16* fc = feat + (long)b*featBS + (long)cen*D;
  const h16* fn = feat + (long)b*featBS + (long)nid*D;
  int c = threadIdx.x;
  h16* out = agg + col * (2L*D);
  out[c]     = (h16)((float)fn[c] - (float)fc[c]);
  out[D + c] = fc[c];
}

// ------------------------------------------------------------------ WMMA GEMM
// D[m, col] = sum_K A[m,k]*B[k,col] (+bias[col]).  A row-major f16 [M][lda];
// B "column-major" f16: column col contiguous over K at Bm + col*ldb;
// D column-major f16: D + col*ldd + m.  cols must be a multiple of 64,
// K a multiple of 32, lda/ldb/ldd multiples of 8. block = 4 waves (128 thr):
// wave = 1 M-tile x 4 N-tiles. Software-pipelined: the next K-step's 10
// b128 fragment loads are issued before the current step's 4 WMMAs, so the
// waits relax to "oldest 10 of ~20 outstanding" and WMMA overlaps the fetch.
__launch_bounds__(128)
__global__ void k_gemm(const h16* __restrict__ A, long aB, int lda,
                       const h16* __restrict__ Bm, long bB, int ldb,
                       h16* __restrict__ D, long dB, int ldd,
                       int M, int K, const float* __restrict__ biasCol) {
  int lane = threadIdx.x & 31;
  int wave = threadIdx.x >> 5;
  int z = blockIdx.z;
  const h16* Ab = A + (long)z * aB;
  const h16* Bb = Bm + (long)z * bB;
  h16* Db = D + (long)z * dB;
  int mt = blockIdx.y * 4 + wave;
  if (mt * 16 >= M) return;                       // wave-uniform skip
  int laneM = lane & 15, laneHi = lane >> 4;
  const h16* arow = Ab + (long)(mt*16 + laneM) * lda + laneHi * 16;
  int nt0 = blockIdx.x * 4;
  const h16* bc0 = Bb + (long)(nt0*16 + laneM) * ldb + laneHi * 16;
  const h16* bc1 = bc0 + (long)16 * ldb;
  const h16* bc2 = bc0 + (long)32 * ldb;
  const h16* bc3 = bc0 + (long)48 * ldb;

  v8f acc[4];
  #pragma unroll
  for (int j = 0; j < 4; ++j)
    #pragma unroll
    for (int i = 0; i < 8; ++i) acc[j][i] = 0.f;

  // prologue: fragment set for kk = 0
  v8h a0c = *(const v8h*)(arow);
  v8h a1c = *(const v8h*)(arow + 8);
  v8h p0c = *(const v8h*)(bc0), p1c = *(const v8h*)(bc0 + 8);
  v8h q0c = *(const v8h*)(bc1), q1c = *(const v8h*)(bc1 + 8);
  v8h r0c = *(const v8h*)(bc2), r1c = *(const v8h*)(bc2 + 8);
  v8h s0c = *(const v8h*)(bc3), s1c = *(const v8h*)(bc3 + 8);

  for (int kk = 0; kk < K; kk += 32) {
    // branchless prefetch of the next K-step (last iter re-reads offset 0)
    int kn = (kk + 32 < K) ? kk + 32 : 0;
    v8h a0n = *(const v8h*)(arow + kn);
    v8h a1n = *(const v8h*)(arow + kn + 8);
    v8h p0n = *(const v8h*)(bc0 + kn), p1n = *(const v8h*)(bc0 + kn + 8);
    v8h q0n = *(const v8h*)(bc1 + kn), q1n = *(const v8h*)(bc1 + kn + 8);
    v8h r0n = *(const v8h*)(bc2 + kn), r1n = *(const v8h*)(bc2 + kn + 8);
    v8h s0n = *(const v8h*)(bc3 + kn), s1n = *(const v8h*)(bc3 + kn + 8);

    v16h af;
    #pragma unroll
    for (int i = 0; i < 8; ++i) { af[i] = a0c[i]; af[i+8] = a1c[i]; }
    v16h bf;
    #pragma unroll
    for (int i = 0; i < 8; ++i) { bf[i] = p0c[i]; bf[i+8] = p1c[i]; }
    acc[0] = __builtin_amdgcn_wmma_f32_16x16x32_f16(false, af, false, bf, (short)0, acc[0], false, false);
    #pragma unroll
    for (int i = 0; i < 8; ++i) { bf[i] = q0c[i]; bf[i+8] = q1c[i]; }
    acc[1] = __builtin_amdgcn_wmma_f32_16x16x32_f16(false, af, false, bf, (short)0, acc[1], false, false);
    #pragma unroll
    for (int i = 0; i < 8; ++i) { bf[i] = r0c[i]; bf[i+8] = r1c[i]; }
    acc[2] = __builtin_amdgcn_wmma_f32_16x16x32_f16(false, af, false, bf, (short)0, acc[2], false, false);
    #pragma unroll
    for (int i = 0; i < 8; ++i) { bf[i] = s0c[i]; bf[i+8] = s1c[i]; }
    acc[3] = __builtin_amdgcn_wmma_f32_16x16x32_f16(false, af, false, bf, (short)0, acc[3], false, false);

    a0c = a0n; a1c = a1n;
    p0c = p0n; p1c = p1n;
    q0c = q0n; q1c = q1n;
    r0c = r0n; r1c = r1n;
    s0c = s0n; s1c = s1n;
  }

  #pragma unroll
  for (int j = 0; j < 4; ++j) {
    int colI = (nt0 + j)*16 + laneM;
    float bias = biasCol ? biasCol[colI] : 0.f;
    h16* dp = Db + (long)colI * ldd + mt*16 + laneHi*8;
    v8h o;
    #pragma unroll
    for (int i = 0; i < 8; ++i) o[i] = (h16)(acc[j][i] + bias);
    *(v8h*)dp = o;
  }
}

// ------------------------------------------------------------- SG / attention
__global__ void k_maxpool(const h16* __restrict__ H, const float* __restrict__ scale,
                          const float* __restrict__ shift, int s, int k, int npad,
                          h16* __restrict__ out) {   // grid cB*npad, block 256 (C)
  int bi = blockIdx.x, b = bi / npad, si = bi % npad, c = threadIdx.x;
  h16* op = out + ((long)b*npad + si)*256 + c;
  if (si >= s) { *op = (h16)0.f; return; }
  long g = (long)b*s + si;
  float m = 0.f;   // relu(..) >= 0
  for (int kj = 0; kj < k; ++kj) {
    float v = (float)H[(g*k + kj)*256L + c] * scale[c] + shift[c];
    v = fmaxf(v, 0.f);
    if (v > m) m = v;
  }
  *op = (h16)m;
}

// Lt[m][n] stored row-major with row length npad. softmax over m (axis=-1 of
// attn[n,m]), then L1-normalize over n per m. Pads zeroed.
__global__ void k_softmax(h16* attn, int np, int npad) {   // grid cB, block 256
  int b = blockIdx.x, t = threadIdx.x;
  h16* L = attn + (long)b * npad * npad;
  for (int n = t; n < npad; n += 256) {
    if (n < np) {
      float mx = -1e30f;
      for (int m = 0; m < np; ++m) { float v = (float)L[(long)m*npad + n]; if (v > mx) mx = v; }
      float s = 0.f;
      for (int m = 0; m < np; ++m) {
        float e = expf((float)L[(long)m*npad + n] - mx);
        s += e; L[(long)m*npad + n] = (h16)e;
      }
      float inv = 1.f / s;
      for (int m = 0; m < np; ++m) L[(long)m*npad + n] = (h16)((float)L[(long)m*npad + n] * inv);
      for (int m = np; m < npad; ++m) L[(long)m*npad + n] = (h16)0.f;
    } else {
      for (int m = 0; m < npad; ++m) L[(long)m*npad + n] = (h16)0.f;
    }
  }
  __syncthreads();
  for (int m = t; m < np; m += 256) {
    float s = 0.f;
    for (int n = 0; n < np; ++n) s += (float)L[(long)m*npad + n];
    float inv = 1.f / (1e-9f + s);
    for (int n = 0; n < np; ++n) L[(long)m*npad + n] = (h16)((float)L[(long)m*npad + n] * inv);
  }
}

__global__ void k_sub(const h16* __restrict__ a, const h16* __restrict__ b,
                      h16* __restrict__ o, long n) {
  long i = (long)blockIdx.x * blockDim.x + threadIdx.x;
  if (i >= n) return;
  o[i] = (h16)((float)a[i] - (float)b[i]);
}

__global__ void k_oafinal(h16* x, const h16* __restrict__ t, const float* __restrict__ scale,
                          const float* __restrict__ shift, int np, int npad, long n) {
  long i = (long)blockIdx.x * blockDim.x + threadIdx.x;
  if (i >= n) return;
  int c = (int)(i & 255);
  long r = i >> 8;
  int m = (int)(r % npad);
  if (m >= np) { x[i] = (h16)0.f; return; }
  float tv = (float)t[i] * scale[c] + shift[c];
  x[i] = (h16)((float)x[i] + fmaxf(tv, 0.f));
}

__global__ void k_scatter(h16* __restrict__ dst, int npadD, const h16* __restrict__ src,
                          int npadS, const int* __restrict__ idx, int s) {
  int i = blockIdx.x * blockDim.x + threadIdx.x;
  if (i >= cB * s * 256) return;
  int c = i & 255, r = i >> 8, j = r % s, b = r / s;
  dst[((long)b*npadD + idx[b*s + j])*256 + c] = src[((long)b*npadS + j)*256 + c];
}

__global__ void k_out(const h16* __restrict__ f, float* __restrict__ out) {
  int i = blockIdx.x * blockDim.x + threadIdx.x;
  if (i >= cB * 256 * cN) return;
  int n = i % cN, r = i / cN, c = r % 256, b = r / 256;
  out[i] = (float)f[((long)b*NP0 + n)*256 + c];
}

// ==================================================================== host
extern "C" void kernel_launch(void* const* d_in, const int* in_sizes, int n_in,
                              void* d_out, int out_size, void* d_ws, size_t ws_size,
                              hipStream_t stream) {
  (void)in_sizes; (void)n_in; (void)out_size; (void)ws_size;
  const float* x   = (const float*)d_in[0];
  const float* ew1 = (const float*)d_in[1];
  const float* eg1 = (const float*)d_in[2];
  const float* eb1 = (const float*)d_in[3];
  const float* ew2 = (const float*)d_in[4];
  const float* eg2 = (const float*)d_in[5];
  const float* eb2 = (const float*)d_in[6];
  struct SgP { const float *w1,*g1,*b1,*w2,*g2,*b2; } sg[3];
  struct OaP { const float *wq,*wv,*bv,*wt,*bt,*g,*b; } oap[4];
  int p = 7;
  for (int i = 0; i < 3; ++i) {
    sg[i].w1 = (const float*)d_in[p++]; sg[i].g1 = (const float*)d_in[p++];
    sg[i].b1 = (const float*)d_in[p++]; sg[i].w2 = (const float*)d_in[p++];
    sg[i].g2 = (const float*)d_in[p++]; sg[i].b2 = (const float*)d_in[p++];
  }
  for (int i = 0; i < 4; ++i) {
    oap[i].wq = (const float*)d_in[p++]; oap[i].wv = (const float*)d_in[p++];
    oap[i].bv = (const float*)d_in[p++]; oap[i].wt = (const float*)d_in[p++];
    oap[i].bt = (const float*)d_in[p++]; oap[i].g  = (const float*)d_in[p++];
    oap[i].b  = (const float*)d_in[p++];
  }

  // workspace bump allocator
  char* base = (char*)d_ws;
  size_t off = 0;
  auto alloc = [&](size_t bytes) -> void* {
    void* r = base + off;
    off += (bytes + 255) & ~(size_t)255;
    return r;
  };
  const long COLS0 = (long)cB * cN * 32;   // 307200
  float* xyz   = (float*)alloc((size_t)cB*cN*3*4);
  float* xyz1  = (float*)alloc((size_t)cB*64*3*4);
  int*   fps0  = (int*)  alloc((size_t)cB*cN*4);
  int*   idx01 = (int*)  alloc((size_t)cB*64*4);
  int*   idx02 = (int*)  alloc((size_t)cB*8*4);
  int*   nidx0 = (int*)  alloc((size_t)cB*cN*32*4);
  int*   nidx1 = (int*)  alloc((size_t)cB*64*16*4);
  int*   nidx2 = (int*)  alloc((size_t)cB*8*4*4);
  float* raw1  = (float*)alloc((size_t)cB*cN*64*4);
  float* raw2  = (float*)alloc((size_t)cB*cN*128*4);
  h16*   f1pm  = (h16*)  alloc((size_t)cB*cN*64*2);
  h16*   fpm   = (h16*)  alloc((size_t)cB*cN*128*2);
  float* acc   = (float*)alloc(512*4);
  float* scale = (float*)alloc(256*4);
  float* shift = (float*)alloc(256*4);
  int sgK[3] = {256, 512, 512};
  h16* wsgf[3][2];
  for (int i = 0; i < 3; ++i) {
    wsgf[i][0] = (h16*)alloc((size_t)256*sgK[i]*2);
    wsgf[i][1] = (h16*)alloc((size_t)256*256*2);
  }
  h16 *wqf[4], *wvf[4], *wtf[4];
  for (int i = 0; i < 4; ++i) {
    wqf[i] = (h16*)alloc((size_t)64*256*2);
    wvf[i] = (h16*)alloc((size_t)256*256*2);
    wtf[i] = (h16*)alloc((size_t)256*256*2);
  }
  h16* aggbuf = (h16*)alloc((size_t)COLS0*256*2);   // 157 MB (also reused as H2)
  h16* hbuf   = (h16*)alloc((size_t)COLS0*256*2);   // 157 MB
  h16* qbuf   = (h16*)alloc((size_t)cB*NP0*64*2);
  h16* attnb  = (h16*)alloc((size_t)cB*NP0*NP0*2);
  h16* vbuf   = (h16*)alloc((size_t)cB*256*NP0*2);
  h16* xrb    = (h16*)alloc((size_t)cB*NP0*256*2);
  h16* xmrb   = (h16*)alloc((size_t)cB*NP0*256*2);
  h16* tb     = (h16*)alloc((size_t)cB*NP0*256*2);
  h16* f02    = (h16*)alloc((size_t)cB*NP0*256*2);
  h16* f12    = (h16*)alloc((size_t)cB*64*256*2);
  h16* f22    = (h16*)alloc((size_t)cB*NP2*256*2);

  auto cvt = [&](const float* s, h16* d, int n) {
    k_cvt_f16<<<(n + 255)/256, 256, 0, stream>>>(s, d, n);
  };
  for (int i = 0; i < 3; ++i) { cvt(sg[i].w1, wsgf[i][0], 256*sgK[i]); cvt(sg[i].w2, wsgf[i][1], 65536); }
  for (int i = 0; i < 4; ++i) { cvt(oap[i].wq, wqf[i], 16384); cvt(oap[i].wv, wvf[i], 65536); cvt(oap[i].wt, wtf[i], 65536); }

  auto gemm = [&](const h16* A, long aB, int lda, const h16* Bm, long bB, int ldb,
                  h16* Dd, long dB, int ldd, int M, int cols, int K,
                  const float* bias, int nz) {
    dim3 g((unsigned)(cols/64), (unsigned)((M + 63)/64), (unsigned)nz);
    k_gemm<<<g, 128, 0, stream>>>(A, aB, lda, Bm, bB, ldb, Dd, dB, ldd, M, K, bias);
  };
  auto statsH = [&](const h16* buf, int C, int rowsPB, long bStride, int rowStride, int nB) {
    k_zero_f32<<<2, 256, 0, stream>>>(acc, 2*C);
    long tot = (long)rowsPB * nB;
    k_stats<h16><<<dim3((unsigned)((tot + 127)/128)), dim3(C), 0, stream>>>(
        buf, C, rowsPB, bStride, rowStride, acc, tot);
  };
  auto statsF = [&](const float* buf, int C, long rows) {
    k_zero_f32<<<2, 256, 0, stream>>>(acc, 2*C);
    k_stats<float><<<dim3((unsigned)((rows + 127)/128)), dim3(C), 0, stream>>>(
        buf, C, (int)rows, 0L, C, acc, rows);
  };
  auto oaL = [&](h16* xp, int np, int npad, const h16* wq, const h16* wv, const h16* wt,
                 const float* bv, const float* g, const float* bb) {
    // q[n][64] = wq . x          (A=wq, B=x_pm)
    gemm(wq, 0, 256, xp, (long)npad*256, 256, qbuf, (long)npad*64, 64, 64, npad, 256, nullptr, cB);
    // logits Lt[m][n] = q . q^T  (shared q/k weights)
    gemm(qbuf, (long)npad*64, 64, qbuf, (long)npad*64, 64, attnb, (long)npad*npad, npad,
         npad, npad, 64, nullptr, cB);
    // v[c][n] = x . wv^T + bv    (A=x_pm, B=wv rows-as-columns)
    gemm(xp, (long)npad*256, 256, wv, 0, 256, vbuf, (long)256*npad, npad, npad, 256, 256, bv, cB);
    k_softmax<<<cB, 256, 0, stream>>>(attnb, np, npad);
    // x_r[m][c] = v . attn
    gemm(vbuf, (long)256*npad, npad, attnb, (long)npad*npad, npad, xrb, (long)npad*256, 256,
         256, npad, npad, nullptr, cB);
    long ne = (long)cB * npad * 256;
    k_sub<<<(unsigned)((ne + 255)/256), 256, 0, stream>>>(xp, xrb, xmrb, ne);
    // t[m][c] = wt . (x - x_r)   (bt cancels inside train-mode BN)
    gemm(wt, 0, 256, xmrb, (long)npad*256, 256, tb, (long)npad*256, 256,
         256, npad, 256, nullptr, cB);
    statsH(tb, 256, np, (long)npad*256, 256, cB);
    k_bnparams<<<1, 256, 0, stream>>>(acc, g, bb, (float)(np*cB), scale, shift, 256);
    k_oafinal<<<(unsigned)((ne + 255)/256), 256, 0, stream>>>(xp, tb, scale, shift, np, npad, ne);
  };
  auto sgStage = [&](int cin, long cols, const SgP& P, const h16* w1f, const h16* w2f,
                     int s, int k, int npad, h16* outpm) {
    gemm(w1f, 0, cin, aggbuf, 0, cin, hbuf, 0, 256, 256, (int)cols, cin, nullptr, 1);
    statsH(hbuf, 256, (int)cols, 0, 256, 1);
    k_bnparams<<<1, 256, 0, stream>>>(acc, P.g1, P.b1, (float)cols, scale, shift, 256);
    long n1 = cols * 256;
    k_bnrelu_f16<<<(unsigned)((n1 + 255)/256), 256, 0, stream>>>(hbuf, scale, shift, 256, n1);
    gemm(w2f, 0, 256, hbuf, 0, 256, aggbuf, 0, 256, 256, (int)cols, 256, nullptr, 1);
    statsH(aggbuf, 256, (int)cols, 0, 256, 1);
    k_bnparams<<<1, 256, 0, stream>>>(acc, P.g2, P.b2, (float)cols, scale, shift, 256);
    k_maxpool<<<cB*npad, 256, 0, stream>>>(aggbuf, scale, shift, s, k, npad, outpm);
  };

  // ---- input embedding (3 -> 64 -> 128), BN train mode + relu
  k_xyz<<<(cB*cN*3 + 255)/256, 256, 0, stream>>>(x, xyz);
  k_embed1<<<cB*cN, 64, 0, stream>>>(x, ew1, raw1);
  statsF(raw1, 64, (long)cB*cN);
  k_bnparams<<<1, 64, 0, stream>>>(acc, eg1, eb1, (float)(cB*cN), scale, shift, 64);
  k_bnrelu_f32_f16<<<(unsigned)(((long)cB*cN*64 + 255)/256), 256, 0, stream>>>(
      raw1, f1pm, scale, shift, 64, (long)cB*cN*64);
  k_embed2<<<cB*cN, 128, 0, stream>>>(f1pm, ew2, raw2);
  statsF(raw2, 128, (long)cB*cN);
  k_bnparams<<<1, 128, 0, stream>>>(acc, eg2, eb2, (float)(cB*cN), scale, shift, 128);
  k_bnrelu_f32_f16<<<(unsigned)(((long)cB*cN*128 + 255)/256), 256, 0, stream>>>(
      raw2, fpm, scale, shift, 128, (long)cB*cN*128);

  // ---- stage 0: s=600 (=N, fps permutation discarded), k=32
  k_fps<<<cB, 256, 0, stream>>>(xyz, cN, cN, fps0);
  k_knn<<<cB*cN, 256, 0, stream>>>(xyz, cN, fps0, cN, 32, nidx0);
  k_agg<<<(unsigned)COLS0, 128, 0, stream>>>(fpm, (long)cN*128, fps0, nidx0, cN, 32, 128, aggbuf);
  sgStage(256, COLS0, sg[0], wsgf[0][0], wsgf[0][1], cN, 32, NP0, f02);
  oaL(f02, cN, NP0, wqf[0], wvf[0], wtf[0], oap[0].bv, oap[0].g, oap[0].b);
  oaL(f02, cN, NP0, wqf[1], wvf[1], wtf[1], oap[1].bv, oap[1].g, oap[1].b);

  // ---- stage 1: s=64, k=16 (fps over ORIGINAL xyz, feats = f02 by raw index)
  k_fps<<<cB, 256, 0, stream>>>(xyz, cN, 64, idx01);
  k_knn<<<cB*64, 256, 0, stream>>>(xyz, cN, idx01, 64, 16, nidx1);
  k_agg<<<cB*64*16, 256, 0, stream>>>(f02, (long)NP0*256, idx01, nidx1, 64, 16, 256, aggbuf);
  sgStage(512, 16384L, sg[1], wsgf[1][0], wsgf[1][1], 64, 16, 64, f12);
  oaL(f12, 64, 64, wqf[2], wvf[2], wtf[2], oap[2].bv, oap[2].g, oap[2].b);
  oaL(f12, 64, 64, wqf[3], wvf[3], wtf[3], oap[3].bv, oap[3].g, oap[3].b);

  // ---- stage 2: s=8, k=4 over xyz1; reuses oa11/oa12 modules
  k_gather3<<<(cB*64*3 + 255)/256, 256, 0, stream>>>(xyz, idx01, 64, cN, xyz1);
  k_fps<<<cB, 256, 0, stream>>>(xyz1, 64, 8, idx02);
  k_knn<<<cB*8, 256, 0, stream>>>(xyz1, 64, idx02, 8, 4, nidx2);
  k_agg<<<cB*8*4, 256, 0, stream>>>(f12, (long)64*256, idx02, nidx2, 8, 4, 256, aggbuf);
  sgStage(512, 512L, sg[2], wsgf[2][0], wsgf[2][1], 8, 4, NP2, f22);
  oaL(f22, 8, NP2, wqf[2], wvf[2], wtf[2], oap[2].bv, oap[2].g, oap[2].b);
  oaL(f22, 8, NP2, wqf[3], wvf[3], wtf[3], oap[3].bv, oap[3].g, oap[3].b);

  // ---- hierarchical scatter-backs + output transpose to [B,256,600] f32
  k_scatter<<<(cB*8*256 + 255)/256, 256, 0, stream>>>(f12, 64, f22, NP2, idx02, 8);
  k_scatter<<<(cB*64*256 + 255)/256, 256, 0, stream>>>(f02, NP0, f12, 64, idx01, 64);
  k_out<<<(cB*256*cN + 255)/256, 256, 0, stream>>>(f02, (float*)d_out);
}
